// DeltaNet_39642548142577
// MI455X (gfx1250) — matrix operand order repaired and checked by reading, not compiled, not run on hardware
//
#include <hip/hip_runtime.h>

typedef __attribute__((ext_vector_type(16))) _Float16 v16h;
typedef __attribute__((ext_vector_type(8)))  _Float16 v8h;
typedef __attribute__((ext_vector_type(8)))  float    v8f;
typedef __attribute__((ext_vector_type(2)))  float    v2f;
typedef __attribute__((ext_vector_type(4)))  unsigned int u32x4;
typedef __attribute__((ext_vector_type(8)))  int      i32x8;
typedef __attribute__((ext_vector_type(4)))  int      i32x4;

// Problem constants (from reference setup_inputs)
constexpr int Bb = 8, Nn = 1024, Dd = 1024, Hh = 4, Dh = 256;
constexpr int ROWS = Bb * Nn;            // 8192
constexpr float RMS_EPS = 1e-5f;

// ---------------- workspace layout (bytes) ----------------
constexpr size_t OFF_XH   = 0;                                          // f16 x       16MB
constexpr size_t OFF_WQH  = OFF_XH   + (size_t)ROWS * Dd * 2;
constexpr size_t OFF_WKH  = OFF_WQH  + (size_t)Dd * Dd * 2;
constexpr size_t OFF_WVH  = OFF_WKH  + (size_t)Dd * Dd * 2;
constexpr size_t OFF_WOH  = OFF_WVH  + (size_t)Dd * Dd * 2;
constexpr size_t OFF_Q    = OFF_WOH  + (size_t)Dd * Dd * 2;             // f32 Q       32MB
constexpr size_t OFF_K    = OFF_Q    + (size_t)ROWS * Dd * 4;
constexpr size_t OFF_V    = OFF_K    + (size_t)ROWS * Dd * 4;
constexpr size_t OFF_O    = OFF_V    + (size_t)ROWS * Dd * 4;
constexpr size_t OFF_BETA = OFF_O    + (size_t)ROWS * Dd * 4;
constexpr size_t OFF_OH   = OFF_BETA + (size_t)ROWS * Hh * 4;

// ---------------- TDM: 2D tile load Global -> LDS (ISA 8.3/8.4 D# layout) ----------------
// data_size_log: 0=1B,1=2B,2=4B,3=8B. tile_w elems contiguous, tile_h rows, stride elems.
__device__ __forceinline__ void tdm_load_2d(unsigned lds_addr, const void* gptr,
                                            unsigned tile_w, unsigned tile_h,
                                            unsigned stride, unsigned data_size_log,
                                            unsigned tensor_d0, unsigned tensor_d1) {
  unsigned long long ga = (unsigned long long)(uintptr_t)gptr;
  u32x4 g0;
  g0[0] = 1u;                                              // count=1, user descriptor
  g0[1] = lds_addr;                                        // LDS byte offset
  g0[2] = (unsigned)(ga & 0xFFFFFFFFu);                    // global_addr[31:0]
  g0[3] = (unsigned)((ga >> 32) & 0x1FFFFFFu) | (2u << 30); // global_addr[56:32] | type=2
  i32x8 g1;
  g1[0] = (int)(data_size_log << 16);                      // wg_mask=0 | data_size
  g1[1] = (int)((tensor_d0 & 0xFFFFu) << 16);              // atomic_addr=0 | dim0[15:0]
  g1[2] = (int)((tensor_d0 >> 16) | ((tensor_d1 & 0xFFFFu) << 16));
  g1[3] = (int)((tensor_d1 >> 16) | (tile_w << 16));       // dim1[31:16] | tile_dim0
  g1[4] = (int)tile_h;                                     // tile_dim1 | tile_dim2=0
  g1[5] = (int)stride;                                     // dim0_stride[31:0]
  g1[6] = 0;                                               // stride[47:32] | dim1_stride lo
  g1[7] = 0;
  i32x4 gz4 = {};
  i32x8 gz8 = {};
  // 6-arg variant (clang-23 / therock-10.0): (g0, g1, g2, g3, g4, cpol); 2D: groups 2+ unused
  __builtin_amdgcn_tensor_load_to_lds(g0, g1, gz4, gz4, gz8, 0);
}

// ---------------- f32 -> f16 convert ----------------
__global__ __launch_bounds__(256) void cvt_f32_f16(const float* __restrict__ s,
                                                   _Float16* __restrict__ d, int n) {
  for (int i = blockIdx.x * blockDim.x + threadIdx.x; i < n; i += gridDim.x * blockDim.x)
    d[i] = (_Float16)s[i];
}

// ---------------- f16 GEMM: C_f32[M,N] = A_f16[M,K] @ B_f16[K,N] ----------------
// BM=128, BN=64, BK=32; 256 threads = 8 waves. A tile staged by the Tensor Data Mover
// (tensor_load_to_lds, TENSORcnt); B tile staged transposed by the ALU path.
__global__ __launch_bounds__(256) void gemm_f16(const _Float16* __restrict__ A,
                                                const _Float16* __restrict__ B,
                                                float* __restrict__ C,
                                                int M, int N, int K) {
  __shared__ __align__(32) _Float16 sA[128][32];   // 8KB  (TDM destination)
  __shared__ __align__(32) _Float16 sBT[64][32];   // 4KB, B tile transposed
  const int tid  = threadIdx.x;
  const int lane = tid & 31;
  const int wv   = tid >> 5;
  const int lm   = lane & 15;
  const int hi   = (lane < 16) ? 0 : 8;
  const int bm   = blockIdx.x * 128;
  const int bn   = blockIdx.y * 64;
  const unsigned sA_lds = (unsigned)(uintptr_t)&sA[0][0];

  v8f acc[4] = {};
  for (int k0 = 0; k0 < K; k0 += 32) {
    // ---- A tile via TDM: [128 rows x 32 halves], row stride K halves ----
    if (wv == 0)
      tdm_load_2d(sA_lds, A + (size_t)bm * K + k0, 32, 128, (unsigned)K, 1,
                  (unsigned)K, 1u << 20);
    // ---- B tile: coalesced 16B read, transposed LDS write ----
    {
      int krow = tid >> 3, nseg = (tid & 7) * 8;
      const _Float16* bp = B + (size_t)(k0 + krow) * N + bn + nseg;
      if (k0 + 32 < K) __builtin_prefetch(bp + (size_t)32 * N, 0, 1);
      v8h t = *(const v8h*)bp;
#pragma unroll
      for (int j = 0; j < 8; ++j) sBT[nseg + j][krow] = t[j];
    }
    if (wv == 0) __builtin_amdgcn_s_wait_tensorcnt(0);
    __syncthreads();
    // A fragment (contiguous k order: k = 16*(lane/16) + s, matching B)
    const int ab = (lane < 16) ? 0 : 16;
    v8h alo = *(const v8h*)&sA[wv * 16 + lm][ab];
    v8h ahi = *(const v8h*)&sA[wv * 16 + lm][ab + 8];
    v16h a;
#pragma unroll
    for (int j = 0; j < 8; ++j) { a[j] = alo[j]; a[8 + j] = ahi[j]; }
    const int bb = (lane < 16) ? 0 : 16;
#pragma unroll
    for (int ct = 0; ct < 4; ++ct) {
      v16h b = *(const v16h*)&sBT[ct * 16 + lm][bb];
      acc[ct] = __builtin_amdgcn_wmma_f32_16x16x32_f16(false, a, false, b,
                                                       (short)0, acc[ct], false, false);
    }
    __syncthreads();
  }
#pragma unroll
  for (int ct = 0; ct < 4; ++ct)
#pragma unroll
    for (int r = 0; r < 8; ++r)
      C[(size_t)(bm + wv * 16 + r + hi) * N + bn + ct * 16 + lm] = acc[ct][r];
}

// ---------------- beta = sigmoid(x @ Wbeta), one wave per (row, h) ----------------
__global__ __launch_bounds__(256) void beta_kernel(const float* __restrict__ x,
                                                   const float* __restrict__ Wb,
                                                   float* __restrict__ beta) {
  int gw = (blockIdx.x * blockDim.x + threadIdx.x) >> 5;
  int lane = threadIdx.x & 31;
  int row = gw >> 2, h = gw & 3;
  float s = 0.f;
  for (int d = lane; d < Dd; d += 32) s += x[(size_t)row * Dd + d] * Wb[d * Hh + h];
#pragma unroll
  for (int o = 16; o; o >>= 1) s += __shfl_xor(s, o, 32);
  if (lane == 0) beta[row * Hh + h] = 1.f / (1.f + __expf(-s));
}

// ---------------- in-place per-head L2 norm, one wave per (row, h) ----------------
__global__ __launch_bounds__(256) void l2norm_head(float* __restrict__ X) {
  int gw = (blockIdx.x * blockDim.x + threadIdx.x) >> 5;
  int lane = threadIdx.x & 31;
  int row = gw >> 2, h = gw & 3;
  float* p = X + (size_t)row * Dd + h * Dh;
  float v[8]; float s = 0.f;
#pragma unroll
  for (int i = 0; i < 8; ++i) { v[i] = p[lane + 32 * i]; s += v[i] * v[i]; }
#pragma unroll
  for (int o = 16; o; o >>= 1) s += __shfl_xor(s, o, 32);
  float inv = rsqrtf(s);
#pragma unroll
  for (int i = 0; i < 8; ++i) p[lane + 32 * i] = v[i] * inv;
}

// ---------------- per-head RMSNorm * g, f32 -> f16, one wave per (row, h) ----------------
__global__ __launch_bounds__(256) void rmsnorm_head(const float* __restrict__ O,
                                                    const float* __restrict__ g,
                                                    _Float16* __restrict__ Oh) {
  int gw = (blockIdx.x * blockDim.x + threadIdx.x) >> 5;
  int lane = threadIdx.x & 31;
  int row = gw >> 2, h = gw & 3;
  const float* p = O + (size_t)row * Dd + h * Dh;
  _Float16* q = Oh + (size_t)row * Dd + h * Dh;
  float v[8]; float s = 0.f;
#pragma unroll
  for (int i = 0; i < 8; ++i) { v[i] = p[lane + 32 * i]; s += v[i] * v[i]; }
#pragma unroll
  for (int o = 16; o; o >>= 1) s += __shfl_xor(s, o, 32);
  float scale = rsqrtf(s * (1.f / Dh) + RMS_EPS);
#pragma unroll
  for (int i = 0; i < 8; ++i)
    q[lane + 32 * i] = (_Float16)(v[i] * scale * g[lane + 32 * i]);
}

// ---------------- chunked delta-rule scan: one block per (b,h), S resident in LDS ----------------
// LDS: S f32[256][256] (256KB) | E f32[32][256] (32KB) | Kc f16[32][256] (16KB) | Gk,Gq f32[32][32]
__global__ __launch_bounds__(256, 1) void delta_kernel(const float* __restrict__ Q,
                                                       const float* __restrict__ Kg,
                                                       const float* __restrict__ V,
                                                       const float* __restrict__ beta,
                                                       float* __restrict__ O) {
  extern __shared__ __align__(16) char smem[];
  float*    S  = (float*)smem;                              // 65536 f32
  float*    E  = (float*)(smem + 262144);                   // 32*256 f32
  _Float16* Kc = (_Float16*)(smem + 262144 + 32768);        // 32*256 f16
  float*    Gk = (float*)(smem + 262144 + 32768 + 16384);   // 32*32
  float*    Gq = Gk + 1024;                                 // 32*32

  const int tid  = threadIdx.x;
  const int lane = tid & 31;
  const int wv   = tid >> 5;
  const int lm   = lane & 15;
  const int base = (lane < 16) ? 0 : 2;   // f32 WMMA K-pair selector
  const int hi   = (lane < 16) ? 0 : 8;   // C-matrix row-half
  const int b    = blockIdx.x >> 2;
  const int h    = blockIdx.x & 3;
  const int brow = b * Nn;
  const int cb   = h * Dh;

  for (int i = tid; i < 65536; i += 256) S[i] = 0.f;
  __syncthreads();

  for (int t0 = 0; t0 < Nn; t0 += 32) {
    // ---- stage K chunk (f16) ----
    {
      int tr = tid >> 3, seg = (tid & 7) * 32;
      const float* src = Kg + (size_t)(brow + t0 + tr) * Dd + cb + seg;
#pragma unroll
      for (int j = 0; j < 32; ++j) Kc[tr * 256 + seg + j] = (_Float16)src[j];
    }
    __syncthreads();

    // ---- phase 1: Pred = K_c@S0 -> E ; QS = Q_c@S0 -> regs ; Grams -> LDS ----
    v8f qs[4];
#pragma unroll
    for (int rh = 0; rh < 2; ++rh) {
      v8f p0 = {}, p1 = {};
      for (int kk = 0; kk < 64; ++kk) {
        int kd = kk * 4 + base;
        v2f a, b0, b1;
        a.x = (float)Kc[(rh * 16 + lm) * 256 + kd];
        a.y = (float)Kc[(rh * 16 + lm) * 256 + kd + 1];
        b0.x = S[kd * 256 + wv * 16 + lm];        b0.y = S[(kd + 1) * 256 + wv * 16 + lm];
        b1.x = S[kd * 256 + (wv + 8) * 16 + lm];  b1.y = S[(kd + 1) * 256 + (wv + 8) * 16 + lm];
        p0 = __builtin_amdgcn_wmma_f32_16x16x4_f32(false, a, false, b0, (short)0, p0, false, false);
        p1 = __builtin_amdgcn_wmma_f32_16x16x4_f32(false, a, false, b1, (short)0, p1, false, false);
      }
#pragma unroll
      for (int r = 0; r < 8; ++r) {
        E[(rh * 16 + r + hi) * 256 + wv * 16 + lm]       = p0[r];
        E[(rh * 16 + r + hi) * 256 + (wv + 8) * 16 + lm] = p1[r];
      }
      v8f q0 = {}, q1 = {};
      const float* qrow = Q + (size_t)(brow + t0 + rh * 16 + lm) * Dd + cb;
      for (int kk = 0; kk < 64; ++kk) {
        int kd = kk * 4 + base;
        v2f a, b0, b1;
        a.x = qrow[kd]; a.y = qrow[kd + 1];
        b0.x = S[kd * 256 + wv * 16 + lm];        b0.y = S[(kd + 1) * 256 + wv * 16 + lm];
        b1.x = S[kd * 256 + (wv + 8) * 16 + lm];  b1.y = S[(kd + 1) * 256 + (wv + 8) * 16 + lm];
        q0 = __builtin_amdgcn_wmma_f32_16x16x4_f32(false, a, false, b0, (short)0, q0, false, false);
        q1 = __builtin_amdgcn_wmma_f32_16x16x4_f32(false, a, false, b1, (short)0, q1, false, false);
      }
      qs[rh * 2 + 0] = q0; qs[rh * 2 + 1] = q1;
    }
    // Gram tiles: waves 0..3 -> Gk (2x2 tiles), waves 4..7 -> Gq
    {
      v8f g = {};
      if (wv < 4) {
        int gr = wv >> 1, gc = wv & 1;
        for (int kk = 0; kk < 64; ++kk) {
          int kd = kk * 4 + base;
          v2f a, bb;
          a.x  = (float)Kc[(gr * 16 + lm) * 256 + kd];
          a.y  = (float)Kc[(gr * 16 + lm) * 256 + kd + 1];
          bb.x = (float)Kc[(gc * 16 + lm) * 256 + kd];
          bb.y = (float)Kc[(gc * 16 + lm) * 256 + kd + 1];
          g = __builtin_amdgcn_wmma_f32_16x16x4_f32(false, a, false, bb, (short)0, g, false, false);
        }
#pragma unroll
        for (int r = 0; r < 8; ++r) Gk[(gr * 16 + r + hi) * 32 + gc * 16 + lm] = g[r];
      } else {
        int gi = wv - 4, gr = gi >> 1, gc = gi & 1;
        const float* qrow = Q + (size_t)(brow + t0 + gr * 16 + lm) * Dd + cb;
        for (int kk = 0; kk < 64; ++kk) {
          int kd = kk * 4 + base;
          v2f a, bb;
          a.x  = qrow[kd]; a.y = qrow[kd + 1];
          bb.x = (float)Kc[(gc * 16 + lm) * 256 + kd];
          bb.y = (float)Kc[(gc * 16 + lm) * 256 + kd + 1];
          g = __builtin_amdgcn_wmma_f32_16x16x4_f32(false, a, false, bb, (short)0, g, false, false);
        }
#pragma unroll
        for (int r = 0; r < 8; ++r) Gq[(gr * 16 + r + hi) * 32 + gc * 16 + lm] = g[r];
      }
    }
    __syncthreads();

    // ---- phase 2: column-private triangular err recurrence (no barriers needed) ----
    {
      const int v = tid;  // owns Dv column v
      float ecol[32];
      for (int t = 0; t < 32; ++t) {
        float acc = E[t * 256 + v];
        for (int s = 0; s < t; ++s) acc += Gk[t * 32 + s] * ecol[s];
        float bta = beta[(brow + t0 + t) * Hh + h];
        float vv  = V[(size_t)(brow + t0 + t) * Dd + cb + v];
        ecol[t] = (vv - acc) * bta;
      }
#pragma unroll
      for (int t = 0; t < 32; ++t) E[t * 256 + v] = ecol[t];
    }
    __syncthreads();

    // ---- phase 3a: O = QS + tril(Gq) @ Err ----
#pragma unroll
    for (int rh = 0; rh < 2; ++rh)
#pragma unroll
      for (int ci = 0; ci < 2; ++ci) {
        int ct = wv + ci * 8;
#pragma unroll
        for (int r = 0; r < 8; ++r) {
          int trow = rh * 16 + r + hi;
          int col  = ct * 16 + lm;
          float o = qs[rh * 2 + ci][r];
          for (int s = 0; s <= trow; ++s) o += Gq[trow * 32 + s] * E[s * 256 + col];
          O[(size_t)(brow + t0 + trow) * Dd + cb + col] = o;
        }
      }

    // ---- phase 3b: S += K_c^T @ Err (rank-32, WMMA), 32 tiles per wave ----
    for (int ti = 0; ti < 32; ++ti) {
      int tau = wv * 32 + ti;
      int di = tau >> 4, vj = tau & 15;
      v8f c;
#pragma unroll
      for (int r = 0; r < 8; ++r) c[r] = S[(di * 16 + r + hi) * 256 + vj * 16 + lm];
#pragma unroll
      for (int kk = 0; kk < 8; ++kk) {
        int tt = kk * 4 + base;
        v2f a, bb;
        a.x  = (float)Kc[tt * 256 + di * 16 + lm];
        a.y  = (float)Kc[(tt + 1) * 256 + di * 16 + lm];
        bb.x = E[tt * 256 + vj * 16 + lm];
        bb.y = E[(tt + 1) * 256 + vj * 16 + lm];
        c = __builtin_amdgcn_wmma_f32_16x16x4_f32(false, a, false, bb, (short)0, c, false, false);
      }
#pragma unroll
      for (int r = 0; r < 8; ++r) S[(di * 16 + r + hi) * 256 + vj * 16 + lm] = c[r];
    }
    __syncthreads();
  }
}

// ---------------- launch ----------------
extern "C" void kernel_launch(void* const* d_in, const int* in_sizes, int n_in,
                              void* d_out, int out_size, void* d_ws, size_t ws_size,
                              hipStream_t stream) {
  (void)in_sizes; (void)n_in; (void)out_size; (void)ws_size;
  const float* x   = (const float*)d_in[0];
  const float* Wq  = (const float*)d_in[1];
  const float* Wk  = (const float*)d_in[2];
  const float* Wv  = (const float*)d_in[3];
  const float* Wb  = (const float*)d_in[4];
  const float* Wo  = (const float*)d_in[5];
  const float* gN  = (const float*)d_in[6];
  char* ws = (char*)d_ws;

  _Float16* xh  = (_Float16*)(ws + OFF_XH);
  _Float16* Wqh = (_Float16*)(ws + OFF_WQH);
  _Float16* Wkh = (_Float16*)(ws + OFF_WKH);
  _Float16* Wvh = (_Float16*)(ws + OFF_WVH);
  _Float16* Woh = (_Float16*)(ws + OFF_WOH);
  float* Qf  = (float*)(ws + OFF_Q);
  float* Kf  = (float*)(ws + OFF_K);
  float* Vf  = (float*)(ws + OFF_V);
  float* Of  = (float*)(ws + OFF_O);
  float* bet = (float*)(ws + OFF_BETA);
  _Float16* Ohh = (_Float16*)(ws + OFF_OH);

  cvt_f32_f16<<<2048, 256, 0, stream>>>(x,  xh,  ROWS * Dd);
  cvt_f32_f16<<<512,  256, 0, stream>>>(Wq, Wqh, Dd * Dd);
  cvt_f32_f16<<<512,  256, 0, stream>>>(Wk, Wkh, Dd * Dd);
  cvt_f32_f16<<<512,  256, 0, stream>>>(Wv, Wvh, Dd * Dd);
  cvt_f32_f16<<<512,  256, 0, stream>>>(Wo, Woh, Dd * Dd);

  dim3 gg(ROWS / 128, Dd / 64);
  gemm_f16<<<gg, 256, 0, stream>>>(xh, Wqh, Qf, ROWS, Dd, Dd);
  gemm_f16<<<gg, 256, 0, stream>>>(xh, Wkh, Kf, ROWS, Dd, Dd);
  gemm_f16<<<gg, 256, 0, stream>>>(xh, Wvh, Vf, ROWS, Dd, Dd);

  beta_kernel<<<ROWS * Hh / 8, 256, 0, stream>>>(x, Wb, bet);
  l2norm_head<<<ROWS * Hh / 8, 256, 0, stream>>>(Qf);
  l2norm_head<<<ROWS * Hh / 8, 256, 0, stream>>>(Kf);

  constexpr size_t DELTA_LDS = 262144 + 32768 + 16384 + 4096 + 4096; // 312KB <= 320KB/WGP
  delta_kernel<<<Bb * Hh, 256, DELTA_LDS, stream>>>(Qf, Kf, Vf, bet, Of);

  rmsnorm_head<<<ROWS * Hh / 8, 256, 0, stream>>>(Of, gN, Ohh);
  gemm_f16<<<gg, 256, 0, stream>>>(Ohh, Woh, (float*)d_out, ROWS, Dd, Dd);
}
